// AttributeBBoxHead_14216341750014
// MI455X (gfx1250) — matrix operand order repaired and checked by reading, not compiled, not run on hardware
//
#include <hip/hip_runtime.h>

typedef __attribute__((ext_vector_type(16))) __bf16 v16bf;
typedef __attribute__((ext_vector_type(8)))  float  v8f;

#define K_DIM   12544
#define KT      (K_DIM / 32)      // 392 K-steps
#define NT      11                // 176 padded columns / 16
#define N_ROIS  8192
// packed B-fragment element count: KT * NT * 32 lanes * 16 halves
#define PACK_TOT (KT * NT * 32 * 16)   // 2,207,744 bf16 elements per copy

// Output region offsets (flat concat: cls, reg, face, colour, motion)
#define OFF_CLS    0
#define OFF_REG    262144      // 8192*32
#define OFF_FACE   1277952     // +8192*124
#define OFF_COL    1302528     // +8192*3
#define OFF_MOT    1359872     // +8192*7

// ---------------------------------------------------------------------------
// Pack the 5 head weight matrices into WMMA B-fragment layout, split into
// bf16 hi + bf16 lo (residual) so 3 bf16 WMMAs reconstruct ~fp32 accuracy.
// Layout: element (kt, nt, lane, h) at ((kt*NT + nt)*32 + lane)*16 + h
// B-fragment map (16-bit B, 32x16):  N = nt*16 + (lane&15),
//                                    K = kt*32 + (lane<16 ? 0 : 8) + (h<8 ? h : h+8)
// ---------------------------------------------------------------------------
__global__ __launch_bounds__(256) void pack_w_kernel(
    const float* __restrict__ Wc, const float* __restrict__ Wr,
    const float* __restrict__ Wf, const float* __restrict__ Wcol,
    const float* __restrict__ Wm,
    __bf16* __restrict__ hi, __bf16* __restrict__ lo)
{
    int idx = blockIdx.x * blockDim.x + threadIdx.x;
    if (idx >= PACK_TOT) return;
    int h    = idx & 15;
    int lane = (idx >> 4) & 31;
    int rest = idx >> 9;
    int nt   = rest % NT;
    int kt   = rest / NT;

    int k = kt * 32 + ((lane < 16) ? 0 : 8) + ((h < 8) ? h : h + 8);
    int n = nt * 16 + (lane & 15);

    float w = 0.0f;
    if      (n < 32)  w = Wc  [(size_t)k * 32  + n];
    else if (n < 156) w = Wr  [(size_t)k * 124 + (n - 32)];
    else if (n < 159) w = Wf  [(size_t)k * 3   + (n - 156)];
    else if (n < 166) w = Wcol[(size_t)k * 7   + (n - 159)];
    else if (n < 168) w = Wm  [(size_t)k * 2   + (n - 166)];

    __bf16 wh = (__bf16)w;
    hi[idx] = wh;
    lo[idx] = (__bf16)(w - (float)wh);
}

// ---------------------------------------------------------------------------
// Fused 5-head GEMM. One wave owns a 16-row M tile and all 11 N tiles.
// A fragments (16x32 bf16) are gathered from fp32 x with the documented
// A-layout (lane<16: K {0..7,16..23}, lane>=16: +8), split hi/lo in regs.
// ---------------------------------------------------------------------------
__global__ __launch_bounds__(128) void attr_head_gemm(
    const float* __restrict__ x,
    const __bf16* __restrict__ Bhi, const __bf16* __restrict__ Blo,
    const float* __restrict__ b_cls, const float* __restrict__ b_reg,
    const float* __restrict__ b_face, const float* __restrict__ b_col,
    const float* __restrict__ b_mot,
    float* __restrict__ out)
{
    const int lane  = threadIdx.x & 31;
    const int wave  = threadIdx.x >> 5;
    const int mtile = blockIdx.x * 4 + wave;          // 16-row tile id
    const int row_a = mtile * 16 + (lane & 15);       // A row fed by this lane
    const int kgrp  = (lane < 16) ? 0 : 8;            // A-layout K sub-block
    const float* xrow = x + (size_t)row_a * K_DIM + kgrp;

    v8f acc[NT] = {};

    const v16bf* bhiv = (const v16bf*)Bhi;
    const v16bf* blov = (const v16bf*)Blo;

    for (int kt = 0; kt < KT; ++kt) {
        const int kb = kt * 32;
        // Two contiguous 32B runs per lane: K rel {0..7} and {16..23} (+kgrp)
        float4 f0 = *(const float4*)(xrow + kb);
        float4 f1 = *(const float4*)(xrow + kb + 4);
        float4 f2 = *(const float4*)(xrow + kb + 16);
        float4 f3 = *(const float4*)(xrow + kb + 20);
        __builtin_prefetch(xrow + kb + 96, 0, 0);     // global_prefetch_b8

        float fv[16] = {f0.x, f0.y, f0.z, f0.w, f1.x, f1.y, f1.z, f1.w,
                        f2.x, f2.y, f2.z, f2.w, f3.x, f3.y, f3.z, f3.w};
        v16bf ah, al;
        #pragma unroll
        for (int h = 0; h < 16; ++h) {
            __bf16 hv = (__bf16)fv[h];
            ah[h] = hv;
            al[h] = (__bf16)(fv[h] - (float)hv);
        }

        const size_t bbase = ((size_t)kt * NT) * 32 + lane;
        #pragma unroll
        for (int nt = 0; nt < NT; ++nt) {
            v16bf bh = bhiv[bbase + (size_t)nt * 32];
            v16bf bl = blov[bbase + (size_t)nt * 32];
            acc[nt] = __builtin_amdgcn_wmma_f32_16x16x32_bf16(
                false, ah, false, bh, (short)0, acc[nt], false, false);
            acc[nt] = __builtin_amdgcn_wmma_f32_16x16x32_bf16(
                false, al, false, bh, (short)0, acc[nt], false, false);
            acc[nt] = __builtin_amdgcn_wmma_f32_16x16x32_bf16(
                false, ah, false, bl, (short)0, acc[nt], false, false);
        }
    }

    // Epilogue: C/D layout => N = lane&15, M = j + (lane>=16 ? 8 : 0)
    const int mrow0 = mtile * 16 + ((lane >> 4) << 3);
    #pragma unroll
    for (int nt = 0; nt < NT; ++nt) {
        const int n = nt * 16 + (lane & 15);
        float bias; size_t base; int dim, rel;
        if      (n < 32)  { bias = b_cls [n];       base = OFF_CLS;  dim = 32;  rel = n;       }
        else if (n < 156) { bias = b_reg [n - 32];  base = OFF_REG;  dim = 124; rel = n - 32;  }
        else if (n < 159) { bias = b_face[n - 156]; base = OFF_FACE; dim = 3;   rel = n - 156; }
        else if (n < 166) { bias = b_col [n - 159]; base = OFF_COL;  dim = 7;   rel = n - 159; }
        else if (n < 168) { bias = b_mot [n - 166]; base = OFF_MOT;  dim = 2;   rel = n - 166; }
        else continue;  // padding columns 168..175
        #pragma unroll
        for (int j = 0; j < 8; ++j) {
            out[base + (size_t)(mrow0 + j) * dim + rel] = acc[nt][j] + bias;
        }
    }
}

extern "C" void kernel_launch(void* const* d_in, const int* in_sizes, int n_in,
                              void* d_out, int out_size, void* d_ws, size_t ws_size,
                              hipStream_t stream) {
    const float* x      = (const float*)d_in[0];
    const float* W_cls  = (const float*)d_in[1];
    const float* b_cls  = (const float*)d_in[2];
    const float* W_reg  = (const float*)d_in[3];
    const float* b_reg  = (const float*)d_in[4];
    const float* W_face = (const float*)d_in[5];
    const float* b_face = (const float*)d_in[6];
    const float* W_col  = (const float*)d_in[7];
    const float* b_col  = (const float*)d_in[8];
    const float* W_mot  = (const float*)d_in[9];
    const float* b_mot  = (const float*)d_in[10];
    float* out = (float*)d_out;

    __bf16* w_hi = (__bf16*)d_ws;            // PACK_TOT bf16 (4.4 MB)
    __bf16* w_lo = w_hi + PACK_TOT;          // PACK_TOT bf16 (4.4 MB)

    pack_w_kernel<<<(PACK_TOT + 255) / 256, 256, 0, stream>>>(
        W_cls, W_reg, W_face, W_col, W_mot, w_hi, w_lo);

    attr_head_gemm<<<N_ROIS / 64, 128, 0, stream>>>(
        x, w_hi, w_lo, b_cls, b_reg, b_face, b_col, b_mot, out);
}
